// T5Layer_38388417691982
// MI455X (gfx1250) — compile-verified, matrix operand order checked by reading, not run on hardware
//
#include <hip/hip_runtime.h>
#include <hip/hip_bf16.h>
#include <cmath>

// ---------------------------------------------------------------------------
// Types for CDNA5 WMMA / TDM (wave32)
// ---------------------------------------------------------------------------
typedef __attribute__((ext_vector_type(16))) __bf16 v16bf;
typedef __attribute__((ext_vector_type(8)))  float  v8f;
typedef __attribute__((ext_vector_type(4)))  unsigned int v4u;
typedef __attribute__((ext_vector_type(8)))  int    v8i;
typedef __attribute__((ext_vector_type(4)))  int    v4i;

#define DEV_INLINE __device__ __forceinline__

DEV_INLINE v8f wmma_bf16(v16bf a, v16bf b, v8f c) {
    // D = A(16x32 bf16) * B(32x16 bf16) + C(16x16 f32)
    return __builtin_amdgcn_wmma_f32_16x16x32_bf16(
        /*neg_a=*/false, a, /*neg_b=*/false, b,
        /*c_mod=*/(short)0, c, /*reuse_a=*/false, /*reuse_b=*/false);
}

DEV_INLINE unsigned short f2bf(float f) {
    unsigned u = __float_as_uint(f);
    unsigned r = u + 0x7fffu + ((u >> 16) & 1u);   // round-to-nearest-even
    return (unsigned short)(r >> 16);
}

// Load a 16-bf16 fragment as two contiguous 128-bit chunks.
DEV_INLINE v16bf load_frag(const unsigned short* p0, const unsigned short* p1) {
    union { uint4 u[2]; v16bf v; } t;
    t.u[0] = *(const uint4*)p0;
    t.u[1] = *(const uint4*)p1;
    return t.v;
}

// 16-bit A/B fragment addressing for 16x16x32 WMMA (per ISA 7.12.2):
// lanes 0-15: row = lane, K chunks {0..7} and {16..23}
// lanes 16-31: row = lane-16, K chunks {8..15} and {24..31}
DEV_INLINE void frag_row_chunks(int lane, int& row, int& c0, int& c1) {
    if (lane < 16) { row = lane;      c0 = 0; c1 = 16; }
    else           { row = lane - 16; c0 = 8; c1 = 24; }
}

DEV_INLINE float gelu_new(float x) {
    const float c = 0.7978845608028654f; // sqrt(2/pi)
    float x3 = x * x * x;
    return 0.5f * x * (1.0f + tanhf(c * (x + 0.044715f * x3)));
}

// ---------------------------------------------------------------------------
// Tensor Data Mover: async 2D bf16 tile load (global, row-major, strided)
// into LDS (row-major, packed). D# layout per CDNA5 ISA ch.8.
// Issue from ONE wave only; follow with s_wait_tensorcnt + workgroup barrier.
// This toolchain exposes the 6-arg builtin:
//   (uint32x4 g0, int32x8 g1, int32x4 g2, int32x4 g3, int32x8 g4, i32 cpol)
// ---------------------------------------------------------------------------
DEV_INLINE void tdm_load_tile_2d_bf16(unsigned lds_off, const void* gaddr,
                                      int width, int rows, int row_stride) {
    unsigned long long ga = (unsigned long long)(size_t)gaddr;
    v4u g0;
    g0[0] = 1u;                                        // count=1, user mode
    g0[1] = lds_off;                                   // LDS byte address
    g0[2] = (unsigned)(ga & 0xffffffffu);              // global_addr[31:0]
    g0[3] = (unsigned)((ga >> 32) & 0x01ffffffu)       // global_addr[56:32]
          | (2u << 30);                                // type = 2 ("image")
    v8i g1;
    g1[0] = (int)(1u << 16);                           // data_size=1 (2 bytes)
    g1[1] = (int)((unsigned)width << 16);              // tensor_dim0[15:0] @ [63:48]
    g1[2] = (int)((unsigned)rows << 16);               // dim0[31:16]=0, tensor_dim1[15:0]
    g1[3] = (int)((unsigned)width << 16);              // dim1[31:16]=0, tile_dim0
    g1[4] = rows;                                      // tile_dim1; tile_dim2=0
    g1[5] = row_stride;                                // tensor_dim0_stride[31:0]
    g1[6] = 0;                                         // stride hi, dim1_stride lo
    g1[7] = 0;
    v4i gz4 = {0, 0, 0, 0};                            // 2-D: groups 2/3 unused
    v8i gz8 = {0, 0, 0, 0, 0, 0, 0, 0};
    __builtin_amdgcn_tensor_load_to_lds(g0, g1, gz4, gz4, gz8, 0);
}

// ---------------------------------------------------------------------------
// Elementwise fp32 -> bf16 conversion
// ---------------------------------------------------------------------------
__global__ void f32_to_bf16_kernel(const float* __restrict__ src,
                                   unsigned short* __restrict__ dst, int n) {
    int i = blockIdx.x * blockDim.x + threadIdx.x;
    if (i < n) dst[i] = f2bf(src[i]);
}

// ---------------------------------------------------------------------------
// T5 relative-position bias table, laid out [H][2S-1]: tab[h*(2S-1) + delta+S-1]
// ---------------------------------------------------------------------------
__global__ void bias_table_kernel(const float* __restrict__ rel_bias,
                                  float* __restrict__ tab, int H, int maxdelta) {
    int idx = blockIdx.x * blockDim.x + threadIdx.x;
    int W = 2 * maxdelta + 1;
    int total = H * W;
    if (idx >= total) return;
    int h = idx / W;
    int d = idx % W - maxdelta;   // delta = k - q
    const int nb = 16, max_exact = 8;     // NUM_BUCKETS/2, nb/2
    int ret = (d > 0) ? nb : 0;
    int n = (d < 0) ? -d : d;
    int bucket;
    if (n < max_exact) {
        bucket = ret + n;
    } else {
        float nf = (float)(n < 1 ? 1 : n);
        int vl = max_exact + (int)(logf(nf / (float)max_exact) /
                                   logf(128.0f / 8.0f) * (float)(nb - max_exact));
        if (vl > nb - 1) vl = nb - 1;
        bucket = ret + vl;
    }
    tab[idx] = rel_bias[bucket * H + h];
}

// ---------------------------------------------------------------------------
// RMSNorm: one block per row, bf16 output
// ---------------------------------------------------------------------------
__global__ __launch_bounds__(256) void rmsnorm_kernel(
    const float* __restrict__ x, const float* __restrict__ w,
    unsigned short* __restrict__ out, int D) {
    __shared__ float red[8];
    int row = blockIdx.x;
    const float* xr = x + row * D;
    float s = 0.f;
    for (int i = threadIdx.x; i < D; i += blockDim.x) { float v = xr[i]; s += v * v; }
    #pragma unroll
    for (int m = 16; m >= 1; m >>= 1) s += __shfl_xor(s, m);
    if ((threadIdx.x & 31) == 0) red[threadIdx.x >> 5] = s;
    __syncthreads();
    if (threadIdx.x < 32) {
        float t = (threadIdx.x < 8) ? red[threadIdx.x] : 0.f;
        #pragma unroll
        for (int m = 4; m >= 1; m >>= 1) t += __shfl_xor(t, m);
        if (threadIdx.x == 0) red[0] = t;
    }
    __syncthreads();
    float rs = rsqrtf(red[0] / (float)D + 1e-6f);
    unsigned short* orow = out + row * D;
    for (int i = threadIdx.x; i < D; i += blockDim.x)
        orow[i] = f2bf(xr[i] * rs * w[i]);
}

// ---------------------------------------------------------------------------
// Tiled bf16 WMMA GEMM. Block tile 64x64, 4 waves, 32x32 per wave, K step 32.
// A tile staged via Tensor Data Mover; B tile staged transposed by VALU.
// EPI 0: C=bf16(A@B)         EPI 1: C=f32(Res + A@B)
// EPI 2: C=bf16(gelu(A@B1) * (A@B2))
// ---------------------------------------------------------------------------
template <int EPI>
__global__ __launch_bounds__(128) void gemm_kernel(
    const unsigned short* __restrict__ A,   // [M x K] bf16 row-major
    const unsigned short* __restrict__ B,   // [K x N] bf16 row-major
    const unsigned short* __restrict__ B2,  // second B for EPI==2
    const float* __restrict__ Res,          // residual for EPI==1
    void* __restrict__ C,
    int M, int N, int K) {
    __shared__ __align__(16) unsigned short As[64 * 32];   // [row][k]
    __shared__ __align__(16) unsigned short Bs[64 * 32];   // transposed: [n][k]
    __shared__ __align__(16) unsigned short Bs2[64 * 32];

    const int tid = threadIdx.x;
    const int wave = tid >> 5, lane = tid & 31;
    const int m0 = blockIdx.y * 64, n0 = blockIdx.x * 64;
    const int wm = (wave >> 1) * 32, wn = (wave & 1) * 32;
    const unsigned lds_as = (unsigned)(size_t)(void*)As;

    v8f acc[2][2] = {};
    v8f acc2[2][2] = {};

    int fr, c0, c1;
    frag_row_chunks(lane, fr, c0, c1);

    for (int k0 = 0; k0 < K; k0 += 32) {
        __syncthreads();
        // Async DMA: A tile (64 rows x 32 cols, row stride K) -> LDS
        if (wave == 0)
            tdm_load_tile_2d_bf16(lds_as, A + m0 * K + k0, 32, 64, K);
        // Prefetch next K tile into L2 while we work on this one.
        if (k0 + 32 < K) {
            if (tid < 64) {
                __builtin_prefetch(A + (m0 + tid) * K + (k0 + 32), 0, 1);
            } else if (tid < 96) {
                __builtin_prefetch(B + (k0 + 32 + (tid - 64)) * N + n0, 0, 1);
            } else if (EPI == 2) {
                __builtin_prefetch(B2 + (k0 + 32 + (tid - 96)) * N + n0, 0, 1);
            }
        }
        // Stage B tile (32x64) transposed into [n][k]
        #pragma unroll
        for (int i = 0; i < 2; i++) {
            int chunk = tid + i * 128;
            int kr = chunk >> 3, cb = (chunk & 7) * 8;
            union { uint4 v; unsigned short s[8]; } u;
            u.v = *(const uint4*)(B + (k0 + kr) * N + (n0 + cb));
            #pragma unroll
            for (int j = 0; j < 8; j++) Bs[(cb + j) * 32 + kr] = u.s[j];
            if (EPI == 2) {
                u.v = *(const uint4*)(B2 + (k0 + kr) * N + (n0 + cb));
                #pragma unroll
                for (int j = 0; j < 8; j++) Bs2[(cb + j) * 32 + kr] = u.s[j];
            }
        }
        if (wave == 0) __builtin_amdgcn_s_wait_tensorcnt(0);
        __syncthreads();

        v16bf a0 = load_frag(&As[(wm + fr) * 32 + c0],      &As[(wm + fr) * 32 + c1]);
        v16bf a1 = load_frag(&As[(wm + 16 + fr) * 32 + c0], &As[(wm + 16 + fr) * 32 + c1]);
        v16bf b0 = load_frag(&Bs[(wn + fr) * 32 + c0],      &Bs[(wn + fr) * 32 + c1]);
        v16bf b1 = load_frag(&Bs[(wn + 16 + fr) * 32 + c0], &Bs[(wn + 16 + fr) * 32 + c1]);

        acc[0][0] = wmma_bf16(a0, b0, acc[0][0]);
        acc[0][1] = wmma_bf16(a0, b1, acc[0][1]);
        acc[1][0] = wmma_bf16(a1, b0, acc[1][0]);
        acc[1][1] = wmma_bf16(a1, b1, acc[1][1]);

        if (EPI == 2) {
            v16bf bb0 = load_frag(&Bs2[(wn + fr) * 32 + c0],      &Bs2[(wn + fr) * 32 + c1]);
            v16bf bb1 = load_frag(&Bs2[(wn + 16 + fr) * 32 + c0], &Bs2[(wn + 16 + fr) * 32 + c1]);
            acc2[0][0] = wmma_bf16(a0, bb0, acc2[0][0]);
            acc2[0][1] = wmma_bf16(a0, bb1, acc2[0][1]);
            acc2[1][0] = wmma_bf16(a1, bb0, acc2[1][0]);
            acc2[1][1] = wmma_bf16(a1, bb1, acc2[1][1]);
        }
    }

    // Epilogue. C 16x16 f32 layout: lanes 0-15 col=lane rows 0..7 (vgpr r),
    // lanes 16-31 col=lane-16 rows 8..15.
    const int row_off = (lane < 16) ? 0 : 8;
    const int col = lane & 15;
    #pragma unroll
    for (int ti = 0; ti < 2; ti++)
        #pragma unroll
        for (int tj = 0; tj < 2; tj++)
            #pragma unroll
            for (int r = 0; r < 8; r++) {
                int gr = m0 + wm + ti * 16 + row_off + r;
                int gc = n0 + wn + tj * 16 + col;
                int idx = gr * N + gc;
                float v = acc[ti][tj][r];
                if (EPI == 0) {
                    ((unsigned short*)C)[idx] = f2bf(v);
                } else if (EPI == 1) {
                    ((float*)C)[idx] = Res[idx] + v;
                } else {
                    ((unsigned short*)C)[idx] = f2bf(gelu_new(v) * acc2[ti][tj][r]);
                }
            }
}

// ---------------------------------------------------------------------------
// Flash attention: grid (S/64, B*H), 128 threads (4 waves, 16 q-rows each).
// K tile staged via Tensor Data Mover; V tile staged transposed by VALU.
// ---------------------------------------------------------------------------
__global__ __launch_bounds__(128) void attn_kernel(
    const unsigned short* __restrict__ Q,    // [B*S, H*64] bf16
    const unsigned short* __restrict__ Km,
    const unsigned short* __restrict__ Vm,
    const float* __restrict__ bias_tab,      // [H][2S-1]
    const float* __restrict__ mask,          // [B, S]
    unsigned short* __restrict__ Ctx,        // [B*S, H*64] bf16
    int S, int B, int H) {
    __shared__ __align__(16) unsigned short Ks[64 * 64];       // [key][d]
    __shared__ __align__(16) unsigned short Vs[64 * 64];       // transposed [d][key]
    __shared__ __align__(16) unsigned short Ps[4][16 * 64];    // per-wave P [q][key]
    __shared__ __align__(16) float biasW[4][80];               // per-wave bias window

    const int qb = blockIdx.x;
    const int bh = blockIdx.y;
    const int b = bh / H, h = bh % H;
    const int tid = threadIdx.x, wave = tid >> 5, lane = tid & 31;
    const int stride = H * 64;
    const unsigned lds_ks = (unsigned)(size_t)(void*)Ks;

    int fr, c0, c1;
    frag_row_chunks(lane, fr, c0, c1);
    const int row_off = (lane < 16) ? 0 : 8;
    const int col = lane & 15;

    const int q0 = qb * 64 + wave * 16;
    // Persistent Q fragments (16 rows x 64 d -> two K=32 fragments)
    const unsigned short* qrow = Q + (b * S + q0 + fr) * stride + h * 64;
    v16bf aq0 = load_frag(qrow + c0,      qrow + c1);
    v16bf aq1 = load_frag(qrow + 32 + c0, qrow + 32 + c1);

    v8f o[4] = {};
    float m_[8], l_[8];
    #pragma unroll
    for (int r = 0; r < 8; r++) { m_[r] = -1e30f; l_[r] = 0.f; }

    const unsigned short* Kbase = Km + b * S * stride + h * 64;
    const unsigned short* Vbase = Vm + b * S * stride + h * 64;
    const float* tabh = bias_tab + h * (2 * S - 1);
    const int tabmax = 2 * S - 2;

    for (int kb = 0; kb < S; kb += 64) {
        __syncthreads();
        // Async DMA: K tile (64 keys x 64 d, row stride = H*64) -> LDS
        if (wave == 0)
            tdm_load_tile_2d_bf16(lds_ks, Kbase + kb * stride, 64, 64, stride);
        // Prefetch next key/value block.
        if (kb + 64 < S) {
            int r = tid & 63;
            const unsigned short* p =
                (tid < 64 ? Kbase : Vbase) + (kb + 64 + r) * stride;
            __builtin_prefetch(p, 0, 1);
        }
        // Stage V transposed [64 d x 64 keys]
        #pragma unroll
        for (int i = 0; i < 4; i++) {
            int chunk = tid + i * 128;          // 512 chunks of 8
            int r = chunk >> 3, cc = (chunk & 7) * 8;
            union { uint4 v; unsigned short s[8]; } u;
            u.v = *(const uint4*)(Vbase + (kb + r) * stride + cc);
            #pragma unroll
            for (int j = 0; j < 8; j++) Vs[(cc + j) * 64 + r] = u.s[j];
        }
        // Stage this wave's bias window: entry j covers (kcol-kb) - (qr-q0) + 15.
        {
            int wbase = kb - q0 - 15 + (S - 1);
            #pragma unroll
            for (int i = lane; i < 80; i += 32) {
                int idx = wbase + i;
                idx = idx < 0 ? 0 : (idx > tabmax ? tabmax : idx);
                biasW[wave][i] = tabh[idx];
            }
        }
        if (wave == 0) __builtin_amdgcn_s_wait_tensorcnt(0);
        __syncthreads();

        // Score tiles: S[16q x 16k] = Q(16x64) * K_tile^T + bias + mask
        v8f st[4];
        #pragma unroll
        for (int kt = 0; kt < 4; kt++) {
            v8f c;
            int kc = kt * 16 + col;             // key within block
            float madd = (1.0f - mask[b * S + kb + kc]) * -10000.0f;
            #pragma unroll
            for (int r = 0; r < 8; r++)
                c[r] = biasW[wave][kc + 15 - row_off - r] + madd;
            v16bf bk0 = load_frag(&Ks[(kt * 16 + fr) * 64 + c0],
                                  &Ks[(kt * 16 + fr) * 64 + c1]);
            v16bf bk1 = load_frag(&Ks[(kt * 16 + fr) * 64 + 32 + c0],
                                  &Ks[(kt * 16 + fr) * 64 + 32 + c1]);
            c = wmma_bf16(aq0, bk0, c);
            c = wmma_bf16(aq1, bk1, c);
            st[kt] = c;
        }

        // Online softmax (rows live in half-waves: reduce over 16 lanes)
        float nm[8], alpha[8];
        #pragma unroll
        for (int r = 0; r < 8; r++) {
            float x = fmaxf(fmaxf(st[0][r], st[1][r]), fmaxf(st[2][r], st[3][r]));
            #pragma unroll
            for (int msk = 8; msk >= 1; msk >>= 1) x = fmaxf(x, __shfl_xor(x, msk));
            nm[r] = fmaxf(m_[r], x);
            alpha[r] = __expf(m_[r] - nm[r]);
            m_[r] = nm[r];
        }
        #pragma unroll
        for (int kt = 0; kt < 4; kt++)
            #pragma unroll
            for (int r = 0; r < 8; r++)
                st[kt][r] = __expf(st[kt][r] - nm[r]);
        #pragma unroll
        for (int r = 0; r < 8; r++) {
            float s = st[0][r] + st[1][r] + st[2][r] + st[3][r];
            #pragma unroll
            for (int msk = 8; msk >= 1; msk >>= 1) s += __shfl_xor(s, msk);
            l_[r] = l_[r] * alpha[r] + s;
        }
        // Rescale O, spill P (C-layout -> LDS row-major -> A fragments)
        #pragma unroll
        for (int t = 0; t < 4; t++)
            #pragma unroll
            for (int r = 0; r < 8; r++) o[t][r] *= alpha[r];
        #pragma unroll
        for (int kt = 0; kt < 4; kt++)
            #pragma unroll
            for (int r = 0; r < 8; r++)
                Ps[wave][(row_off + r) * 64 + kt * 16 + col] = f2bf(st[kt][r]);

        v16bf pa0 = load_frag(&Ps[wave][fr * 64 + c0],      &Ps[wave][fr * 64 + c1]);
        v16bf pa1 = load_frag(&Ps[wave][fr * 64 + 32 + c0], &Ps[wave][fr * 64 + 32 + c1]);
        #pragma unroll
        for (int t = 0; t < 4; t++) {
            v16bf bv0 = load_frag(&Vs[(t * 16 + fr) * 64 + c0],
                                  &Vs[(t * 16 + fr) * 64 + c1]);
            v16bf bv1 = load_frag(&Vs[(t * 16 + fr) * 64 + 32 + c0],
                                  &Vs[(t * 16 + fr) * 64 + 32 + c1]);
            o[t] = wmma_bf16(pa0, bv0, o[t]);
            o[t] = wmma_bf16(pa1, bv1, o[t]);
        }
    }

    #pragma unroll
    for (int r = 0; r < 8; r++) l_[r] = 1.0f / l_[r];
    #pragma unroll
    for (int t = 0; t < 4; t++)
        #pragma unroll
        for (int r = 0; r < 8; r++) {
            int qr = q0 + row_off + r;
            Ctx[(b * S + qr) * stride + h * 64 + t * 16 + col] =
                f2bf(o[t][r] * l_[r]);
        }
}

// ---------------------------------------------------------------------------
// Host launcher
// ---------------------------------------------------------------------------
extern "C" void kernel_launch(void* const* d_in, const int* in_sizes, int n_in,
                              void* d_out, int out_size, void* d_ws, size_t ws_size,
                              hipStream_t stream) {
    const int B = 2, S = 2048, DM = 1024, H = 16, DFF = 4096;
    const int M = B * S;

    const float* hidden_states = (const float*)d_in[0];
    const float* mask          = (const float*)d_in[1];
    const float* ln1_w         = (const float*)d_in[2];
    const float* wq            = (const float*)d_in[3];
    const float* wk            = (const float*)d_in[4];
    const float* wv            = (const float*)d_in[5];
    const float* rel_bias      = (const float*)d_in[6];
    const float* wo            = (const float*)d_in[7];
    const float* ln2_w         = (const float*)d_in[8];
    const float* w1            = (const float*)d_in[9];
    const float* w2            = (const float*)d_in[10];
    const float* w_out         = (const float*)d_in[11];
    float* out = (float*)d_out;

    char* ws = (char*)d_ws;
    size_t off = 0;
    auto alloc = [&](size_t bytes) -> void* {
        void* p = ws + off;
        off = (off + bytes + 255) & ~(size_t)255;
        return p;
    };
    typedef unsigned short us;
    us* wq_bf   = (us*)alloc((size_t)DM * DM * 2);
    us* wk_bf   = (us*)alloc((size_t)DM * DM * 2);
    us* wv_bf   = (us*)alloc((size_t)DM * DM * 2);
    us* wo_bf   = (us*)alloc((size_t)DM * DM * 2);
    us* w1_bf   = (us*)alloc((size_t)DM * DFF * 2);
    us* w2_bf   = (us*)alloc((size_t)DM * DFF * 2);
    us* wout_bf = (us*)alloc((size_t)DFF * DM * 2);
    float* bias_tab = (float*)alloc((size_t)H * (2 * S - 1) * 4);
    us* normed  = (us*)alloc((size_t)M * DM * 2);
    us* qb_     = (us*)alloc((size_t)M * DM * 2);
    us* kb_     = (us*)alloc((size_t)M * DM * 2);
    us* vb_     = (us*)alloc((size_t)M * DM * 2);
    us* ctx     = (us*)alloc((size_t)M * DM * 2);
    float* hidden = (float*)alloc((size_t)M * DM * 4);
    us* normed2 = (us*)alloc((size_t)M * DM * 2);
    us* gated   = (us*)alloc((size_t)M * DFF * 2);
    (void)ws_size; (void)in_sizes; (void)n_in; (void)out_size;

    auto conv = [&](const float* s, us* d, int n) {
        f32_to_bf16_kernel<<<(n + 255) / 256, 256, 0, stream>>>(s, d, n);
    };
    conv(wq, wq_bf, DM * DM);
    conv(wk, wk_bf, DM * DM);
    conv(wv, wv_bf, DM * DM);
    conv(wo, wo_bf, DM * DM);
    conv(w1, w1_bf, DM * DFF);
    conv(w2, w2_bf, DM * DFF);
    conv(w_out, wout_bf, DFF * DM);

    int btot = H * (2 * S - 1);
    bias_table_kernel<<<(btot + 255) / 256, 256, 0, stream>>>(rel_bias, bias_tab, H, S - 1);

    // Pre-LN 1 -> bf16 activations
    rmsnorm_kernel<<<M, 256, 0, stream>>>(hidden_states, ln1_w, normed, DM);

    // QKV projections
    dim3 g_qkv(DM / 64, M / 64);
    gemm_kernel<0><<<g_qkv, 128, 0, stream>>>(normed, wq_bf, nullptr, nullptr, qb_, M, DM, DM);
    gemm_kernel<0><<<g_qkv, 128, 0, stream>>>(normed, wk_bf, nullptr, nullptr, kb_, M, DM, DM);
    gemm_kernel<0><<<g_qkv, 128, 0, stream>>>(normed, wv_bf, nullptr, nullptr, vb_, M, DM, DM);

    // Attention (flash, per (b,h,64-query-block))
    attn_kernel<<<dim3(S / 64, B * H), 128, 0, stream>>>(qb_, kb_, vb_, bias_tab, mask,
                                                         ctx, S, B, H);

    // Output projection + residual -> f32 hidden
    gemm_kernel<1><<<dim3(DM / 64, M / 64), 128, 0, stream>>>(
        ctx, wo_bf, nullptr, hidden_states, hidden, M, DM, DM);

    // Pre-LN 2
    rmsnorm_kernel<<<M, 256, 0, stream>>>(hidden, ln2_w, normed2, DM);

    // Fused gated FFN up-projection: gated = gelu(x@w1) * (x@w2)
    gemm_kernel<2><<<dim3(DFF / 64, M / 64), 128, 0, stream>>>(
        normed2, w1_bf, w2_bf, nullptr, gated, M, DFF, DM);

    // Down-projection + residual -> final output (f32)
    gemm_kernel<1><<<dim3(DM / 64, M / 64), 128, 0, stream>>>(
        gated, wout_bf, nullptr, hidden, out, M, DM, DFF);
}
